// GraphTransformerLayer_79903571575357
// MI455X (gfx1250) — compile-verified, matrix operand order checked by reading, not run on hardware
//
#include <hip/hip_runtime.h>
#include <hip/hip_bf16.h>
#include <math.h>

#define NODES  50000
#define EDGES  1600000
#define INDIM  128
#define HEADS  4
#define OUTC   32
#define HC     128

typedef __bf16 bf16;
typedef __attribute__((ext_vector_type(16))) __bf16 v16bf;
typedef __attribute__((ext_vector_type(8)))  __bf16 v8bf;
typedef __attribute__((ext_vector_type(8)))  float  v8f;

// ---------------- helpers ----------------

__device__ __forceinline__ unsigned encf(float f) {
    unsigned u = __float_as_uint(f);
    return (u & 0x80000000u) ? ~u : (u | 0x80000000u);
}
__device__ __forceinline__ float decf(unsigned u) {
    return (u & 0x80000000u) ? __uint_as_float(u & 0x7FFFFFFFu) : __uint_as_float(~u);
}
__device__ __forceinline__ float elu1(float x) {
    return x > 0.f ? x : (expf(x) - 1.f);
}

// ---------------- weight packing into WMMA B-fragment order ----------------
// B element (k,n) lives in lane = 16*g(k) + (n%16), vector element e(k):
//   k_local in [0,8)+8g   -> e = k_local - 8g        (elements 0..7)
//   k_local in [16,24)+8g -> e = 8 + k_local-16-8g   (elements 8..15)
// Fragment (kt,nt) stored at ((nt*nkt + kt)*32 + lane)*16 contiguous bf16.
__global__ void pack_b_kernel(const float* __restrict__ W, bf16* __restrict__ dst,
                              int K, int Nout) {
    int nkt  = K >> 5;
    int tile = blockIdx.x;            // tile = nt*nkt + kt
    int kt   = tile % nkt;
    int nt   = tile / nkt;
    int lane = threadIdx.x;
    int g    = lane >> 4;
    int n    = nt * 16 + (lane & 15);
    v16bf f;
#pragma unroll
    for (int e = 0; e < 16; ++e) {
        int kl = (e < 8) ? (g * 8 + e) : (16 + g * 8 + (e - 8));
        int kg = kt * 32 + kl;
        f[e] = (bf16)W[(size_t)kg * Nout + n];
    }
    *(v16bf*)(dst + ((size_t)tile * 32 + lane) * 16) = f;
}

// ---------------- generic WMMA GEMM: out = act(A(bf16) @ Bpacked + bias) ----------------
// block = (32,8): wave per (mt, nt) tile; grid = (M/16, nnt/8)
__global__ void gemm_bf16_kernel(const bf16* __restrict__ A, int lda,
                                 const bf16* __restrict__ Bp, int nkt,
                                 const float* __restrict__ bias,
                                 float* __restrict__ outF, bf16* __restrict__ outB,
                                 int ldo, int act) {
    int mt   = blockIdx.x;
    int nt   = blockIdx.y * 8 + threadIdx.y;
    int lane = threadIdx.x;
    int g    = lane >> 4;
    int row  = mt * 16 + (lane & 15);

    const bf16* arow = A + (size_t)row * lda;
    const bf16* bptr = Bp + (size_t)nt * nkt * 32 * 16;

    v8f acc = {};
    for (int kt = 0; kt < nkt; ++kt) {
        int kb = kt * 32;
        v8bf a0 = *(const v8bf*)(arow + kb + g * 8);        // K = kb+8g .. +7
        v8bf a1 = *(const v8bf*)(arow + kb + 16 + g * 8);   // K = kb+16+8g .. +7
        v16bf af;
#pragma unroll
        for (int i = 0; i < 8; ++i) { af[i] = a0[i]; af[i + 8] = a1[i]; }
        v16bf bfr = *(const v16bf*)(bptr + ((size_t)kt * 32 + lane) * 16);
        acc = __builtin_amdgcn_wmma_f32_16x16x32_bf16(false, af, false, bfr,
                                                      (short)0, acc, false, false);
    }

    int   n  = nt * 16 + (lane & 15);
    float bn = bias ? bias[n] : 0.f;
#pragma unroll
    for (int j = 0; j < 8; ++j) {
        int   m = mt * 16 + g * 8 + j;
        float v = acc[j] + bn;
        if (act == 1) v = elu1(v);
        if (outF) outF[(size_t)m * ldo + n] = v;
        if (outB) outB[(size_t)m * ldo + n] = (bf16)v;
    }
}

// ---------------- elementwise convert (optional ELU) fp32 -> bf16 ----------------
__global__ void cvt_bf16_kernel(const float* __restrict__ in, bf16* __restrict__ out,
                                long long n, int act) {
    long long t = (long long)blockIdx.x * blockDim.x + threadIdx.x;
    if (t >= n) return;
    float v = in[t];
    if (act == 1) v = elu1(v);
    out[t] = (bf16)v;
}

// ---------------- init smax/denom ----------------
__global__ void init_attn_kernel(unsigned* __restrict__ smax, float* __restrict__ denom, int n) {
    int t = blockIdx.x * blockDim.x + threadIdx.x;
    if (t < n) { smax[t] = 0u; denom[t] = 0.f; }  // dec(0)=NaN -> only edge-less nodes, never read
}

// ---------------- edge pass 1: scores + segment max ----------------
__global__ void edge_score_kernel(const float* __restrict__ q, const float* __restrict__ k,
                                  const int* __restrict__ src, const int* __restrict__ dst,
                                  float* __restrict__ scores, unsigned* __restrict__ smax) {
    long long t = (long long)blockIdx.x * blockDim.x + threadIdx.x;
    if (t >= (long long)EDGES * HEADS) return;
    int e = (int)(t >> 2), hh = (int)(t & 3);
    int s = src[e], d = dst[e];
    const float4* qp = (const float4*)(q + (size_t)d * HC + hh * OUTC);
    const float4* kp = (const float4*)(k + (size_t)s * HC + hh * OUTC);
    float acc = 0.f;
#pragma unroll
    for (int i = 0; i < 8; ++i) {
        float4 a = qp[i], b = kp[i];
        acc += a.x * b.x + a.y * b.y + a.z * b.z + a.w * b.w;
    }
    acc *= 0.17677669529663687f;   // 1/sqrt(32)
    scores[t] = acc;
    atomicMax(&smax[(size_t)d * HEADS + hh], encf(acc));
}

// ---------------- edge pass 2: p = exp(s - smax[dst]); denom += p ----------------
__global__ void edge_exp_kernel(float* __restrict__ scores, const int* __restrict__ dst,
                                const unsigned* __restrict__ smax, float* __restrict__ denom) {
    long long t = (long long)blockIdx.x * blockDim.x + threadIdx.x;
    if (t >= (long long)EDGES * HEADS) return;
    int e = (int)(t >> 2), hh = (int)(t & 3);
    int d = dst[e];
    float m = decf(smax[(size_t)d * HEADS + hh]);
    float p = expf(scores[t] - m);
    scores[t] = p;
    atomicAdd(&denom[(size_t)d * HEADS + hh], p);
}

// ---------------- edge pass 3: a[dst] += (p/denom) * v[src] ----------------
__global__ void edge_agg_kernel(const float* __restrict__ scores, const float* __restrict__ vmat,
                                const int* __restrict__ src, const int* __restrict__ dst,
                                const float* __restrict__ denom, float* __restrict__ a) {
    long long t = (long long)blockIdx.x * blockDim.x + threadIdx.x;
    if (t >= (long long)EDGES * HEADS) return;
    int e = (int)(t >> 2), hh = (int)(t & 3);
    int s = src[e], d = dst[e];
    float alpha = scores[t] / (denom[(size_t)d * HEADS + hh] + 1e-16f);
    const float4* vp = (const float4*)(vmat + (size_t)s * HC + hh * OUTC);
    float* ap = a + (size_t)d * HC + hh * OUTC;
#pragma unroll
    for (int i = 0; i < 8; ++i) {
        float4 vv = vp[i];
        atomicAdd(ap + 4 * i + 0, alpha * vv.x);
        atomicAdd(ap + 4 * i + 1, alpha * vv.y);
        atomicAdd(ap + 4 * i + 2, alpha * vv.z);
        atomicAdd(ap + 4 * i + 3, alpha * vv.w);
    }
}

// ---------------- LayerNorm over 128 cols: wave per row ----------------
// block = (32,8); y = LN(xa+xb)*g+b -> outF (fp32), optional outB (bf16)
__global__ void layernorm_kernel(const float* __restrict__ xa, const float* __restrict__ xb,
                                 const float* __restrict__ gw, const float* __restrict__ bw,
                                 float* __restrict__ outF, bf16* __restrict__ outB) {
    int row  = blockIdx.x * 8 + threadIdx.y;
    int lane = threadIdx.x;
    const float4 a4 = *(const float4*)(xa + (size_t)row * INDIM + lane * 4);
    const float4 b4 = *(const float4*)(xb + (size_t)row * INDIM + lane * 4);
    float x0 = a4.x + b4.x, x1 = a4.y + b4.y, x2 = a4.z + b4.z, x3 = a4.w + b4.w;

    float s = x0 + x1 + x2 + x3;
#pragma unroll
    for (int o = 16; o > 0; o >>= 1) s += __shfl_xor(s, o, 32);
    float mean = s * (1.f / 128.f);

    float c0 = x0 - mean, c1 = x1 - mean, c2 = x2 - mean, c3 = x3 - mean;
    float sq = c0 * c0 + c1 * c1 + c2 * c2 + c3 * c3;
#pragma unroll
    for (int o = 16; o > 0; o >>= 1) sq += __shfl_xor(sq, o, 32);
    float inv = rsqrtf(sq * (1.f / 128.f) + 1e-5f);

    float c[4] = {c0, c1, c2, c3};
#pragma unroll
    for (int i = 0; i < 4; ++i) {
        int   col = lane * 4 + i;
        float y   = c[i] * inv * gw[col] + bw[col];
        outF[(size_t)row * INDIM + col] = y;
        if (outB) outB[(size_t)row * INDIM + col] = (bf16)y;
    }
}

// ---------------- host orchestration ----------------

extern "C" void kernel_launch(void* const* d_in, const int* in_sizes, int n_in,
                              void* d_out, int out_size, void* d_ws, size_t ws_size,
                              hipStream_t stream) {
    const float* h     = (const float*)d_in[0];
    const int*   eidx  = (const int*)d_in[1];
    const float* Wq    = (const float*)d_in[2];
    const float* bq    = (const float*)d_in[3];
    const float* Wk    = (const float*)d_in[4];
    const float* bk    = (const float*)d_in[5];
    const float* Wv    = (const float*)d_in[6];
    const float* bv    = (const float*)d_in[7];
    const float* Wskip = (const float*)d_in[8];
    const float* bskip = (const float*)d_in[9];
    const float* WO    = (const float*)d_in[10];
    const float* bO    = (const float*)d_in[11];
    const float* W1    = (const float*)d_in[12];
    const float* b1    = (const float*)d_in[13];
    const float* W2    = (const float*)d_in[14];
    const float* b2    = (const float*)d_in[15];
    const float* ln1g  = (const float*)d_in[16];
    const float* ln1b  = (const float*)d_in[17];
    const float* ln2g  = (const float*)d_in[18];
    const float* ln2b  = (const float*)d_in[19];
    const int* srcI = eidx;          // edge_index[0]
    const int* dstI = eidx + EDGES;  // edge_index[1]

    // workspace carve (256B aligned)
    char* w = (char*)d_ws;
    auto carve = [&](size_t bytes) -> char* {
        char* p = w; w += (bytes + 255) & ~(size_t)255; return p;
    };
    bf16*     h_bf   = (bf16*)carve((size_t)NODES * INDIM * sizeof(bf16)); // reused as elu(a)_bf16
    float*    qbuf   = (float*)carve((size_t)NODES * HC * sizeof(float));  // reused as a_trans
    float*    kbuf   = (float*)carve((size_t)NODES * HC * sizeof(float));  // reused as ffn
    float*    vbuf   = (float*)carve((size_t)NODES * HC * sizeof(float));  // reused as h1
    float*    abuf   = (float*)carve((size_t)NODES * HC * sizeof(float));
    float*    scores = (float*)carve((size_t)EDGES * HEADS * sizeof(float)); // reused as mid_bf16
    unsigned* smax   = (unsigned*)carve((size_t)NODES * HEADS * sizeof(unsigned));
    float*    denom  = (float*)carve((size_t)NODES * HEADS * sizeof(float));
    bf16*     h1_bf  = (bf16*)carve((size_t)NODES * INDIM * sizeof(bf16));
    bf16*     pbQ    = (bf16*)carve((size_t)INDIM * HC * sizeof(bf16));
    bf16*     pbK    = (bf16*)carve((size_t)INDIM * HC * sizeof(bf16));
    bf16*     pbV    = (bf16*)carve((size_t)INDIM * HC * sizeof(bf16));
    bf16*     pbS    = (bf16*)carve((size_t)INDIM * HC * sizeof(bf16));
    bf16*     pbO    = (bf16*)carve((size_t)HC * INDIM * sizeof(bf16));
    bf16*     pbW1   = (bf16*)carve((size_t)INDIM * 2 * INDIM * sizeof(bf16));
    bf16*     pbW2   = (bf16*)carve((size_t)2 * INDIM * INDIM * sizeof(bf16));
    bf16*     mid_bf = (bf16*)scores;  // alias: free after edge phase
    float*    atrans = qbuf;
    float*    h1     = vbuf;
    float*    ffn    = kbuf;

    const int MT = NODES / 16;  // 3125 row tiles
    dim3 gblk(32, 8);

    // 1) pack weights to WMMA fragment order
    hipLaunchKernelGGL(pack_b_kernel, dim3(4 * 8),  dim3(32), 0, stream, Wq,    pbQ,  128, 128);
    hipLaunchKernelGGL(pack_b_kernel, dim3(4 * 8),  dim3(32), 0, stream, Wk,    pbK,  128, 128);
    hipLaunchKernelGGL(pack_b_kernel, dim3(4 * 8),  dim3(32), 0, stream, Wv,    pbV,  128, 128);
    hipLaunchKernelGGL(pack_b_kernel, dim3(4 * 8),  dim3(32), 0, stream, Wskip, pbS,  128, 128);
    hipLaunchKernelGGL(pack_b_kernel, dim3(4 * 8),  dim3(32), 0, stream, WO,    pbO,  128, 128);
    hipLaunchKernelGGL(pack_b_kernel, dim3(4 * 16), dim3(32), 0, stream, W1,    pbW1, 128, 256);
    hipLaunchKernelGGL(pack_b_kernel, dim3(8 * 8),  dim3(32), 0, stream, W2,    pbW2, 256, 128);

    // 2) h -> bf16
    long long nh = (long long)NODES * INDIM;
    hipLaunchKernelGGL(cvt_bf16_kernel, dim3((unsigned)((nh + 255) / 256)), dim3(256), 0, stream,
                       h, h_bf, nh, 0);

    // 3) init attention accumulators
    int na = NODES * HEADS;
    hipLaunchKernelGGL(init_attn_kernel, dim3((na + 255) / 256), dim3(256), 0, stream,
                       smax, denom, na);

    // 4) q/k/v/skip projections (WMMA)
    hipLaunchKernelGGL(gemm_bf16_kernel, dim3(MT, 1), gblk, 0, stream,
                       h_bf, 128, pbQ, 4, bq,    qbuf, (bf16*)nullptr, 128, 0);
    hipLaunchKernelGGL(gemm_bf16_kernel, dim3(MT, 1), gblk, 0, stream,
                       h_bf, 128, pbK, 4, bk,    kbuf, (bf16*)nullptr, 128, 0);
    hipLaunchKernelGGL(gemm_bf16_kernel, dim3(MT, 1), gblk, 0, stream,
                       h_bf, 128, pbV, 4, bv,    vbuf, (bf16*)nullptr, 128, 0);
    hipLaunchKernelGGL(gemm_bf16_kernel, dim3(MT, 1), gblk, 0, stream,
                       h_bf, 128, pbS, 4, bskip, abuf, (bf16*)nullptr, 128, 0);

    // 5) edge phase: scores -> softmax -> scatter-aggregate into abuf
    long long et = (long long)EDGES * HEADS;
    unsigned eblocks = (unsigned)((et + 255) / 256);
    hipLaunchKernelGGL(edge_score_kernel, dim3(eblocks), dim3(256), 0, stream,
                       qbuf, kbuf, srcI, dstI, scores, smax);
    hipLaunchKernelGGL(edge_exp_kernel, dim3(eblocks), dim3(256), 0, stream,
                       scores, dstI, smax, denom);
    hipLaunchKernelGGL(edge_agg_kernel, dim3(eblocks), dim3(256), 0, stream,
                       scores, vbuf, srcI, dstI, denom, abuf);

    // 6) elu(a) -> bf16 (reuse h_bf)
    hipLaunchKernelGGL(cvt_bf16_kernel, dim3((unsigned)((nh + 255) / 256)), dim3(256), 0, stream,
                       abuf, h_bf, nh, 1);

    // 7) a_trans = elu(a) @ WO + bO (WMMA)
    hipLaunchKernelGGL(gemm_bf16_kernel, dim3(MT, 1), gblk, 0, stream,
                       h_bf, 128, pbO, 4, bO, atrans, (bf16*)nullptr, 128, 0);

    // 8) h1 = LN1(a_trans + h); also bf16 copy for FFN
    hipLaunchKernelGGL(layernorm_kernel, dim3(NODES / 8), gblk, 0, stream,
                       atrans, h, ln1g, ln1b, h1, h1_bf);

    // 9) mid = elu(h1 @ W1 + b1) -> bf16 only (WMMA, Nout=256)
    hipLaunchKernelGGL(gemm_bf16_kernel, dim3(MT, 2), gblk, 0, stream,
                       h1_bf, 128, pbW1, 4, b1, (float*)nullptr, mid_bf, 256, 1);

    // 10) ffn = mid @ W2 + b2 (WMMA, K=256)
    hipLaunchKernelGGL(gemm_bf16_kernel, dim3(MT, 1), gblk, 0, stream,
                       mid_bf, 256, pbW2, 8, b2, ffn, (bf16*)nullptr, 128, 0);

    // 11) out = LN2(ffn + h1)
    hipLaunchKernelGGL(layernorm_kernel, dim3(NODES / 8), gblk, 0, stream,
                       ffn, h1, ln2g, ln2b, (float*)d_out, (bf16*)nullptr);
}
